// ImprovedSeq2SeqTransformerClassifier_11424613007500
// MI455X (gfx1250) — matrix operand order repaired and checked by reading, not compiled, not run on hardware
//
#include <hip/hip_runtime.h>
#include <hip/hip_bf16.h>
#include <cstddef>
#include <cstdint>

// ---------------- problem dims (from reference) ----------------
#define B_ 4
#define S_ 1024
#define IN_ 1280
#define D_ 512
#define H_ 8
#define DH_ 64
#define LH_ 256
#define FF_ 2048
#define L_ 2

typedef __attribute__((ext_vector_type(16))) __bf16       bf16x16;
typedef __attribute__((ext_vector_type(8)))  float        f32x8;
typedef __attribute__((ext_vector_type(4)))  float        f32x4;
typedef __attribute__((ext_vector_type(4)))  unsigned int u32x4;

#define WMMA(a, b, c) \
  __builtin_amdgcn_wmma_f32_16x16x32_bf16(false, (a), false, (b), (short)0, (c), false, false)

// ---------------- WMMA fragment loaders (ISA 7.12.2 layouts) ----------------
// A: 16x32 bf16. lane(0-15): M=lane, K=[k0..k0+7]+[k0+16..k0+23];
//                lane(16-31): M=lane-16, K=[k0+8..k0+15]+[k0+24..k0+31]
__device__ __forceinline__ bf16x16 fragA(const __bf16* base, int row, int rstride,
                                         int k0, int lane) {
  int hi = (lane >> 4) & 1;
  const __bf16* p = base + (size_t)row * rstride + k0 + hi * 8;
  union { bf16x16 v; u32x4 q[2]; } u;
  u.q[0] = *(const u32x4*)(p);
  u.q[1] = *(const u32x4*)(p + 16);
  return u.v;
}
// B: 32x16 bf16 tile of B = W^T from row-major W[N,K]. lanes 0-15: K=[k0..k0+7]+[k0+16..k0+23]
// (cols = lane), lanes 16-31: K=[k0+8..k0+15]+[k0+24..k0+31] (same k order as fragA).
__device__ __forceinline__ bf16x16 fragB(const __bf16* base, int nrow, int rstride,
                                         int k0, int lane) {
  int hi = (lane >> 4) & 1;
  const __bf16* p = base + (size_t)nrow * rstride + k0 + hi * 8;
  union { bf16x16 v; u32x4 q[2]; } u;
  u.q[0] = *(const u32x4*)(p);
  u.q[1] = *(const u32x4*)(p + 16);
  return u.v;
}

__device__ __forceinline__ float sigm(float x) { return 1.0f / (1.0f + expf(-x)); }

// ---------------- generic GEMM: C[M,N] = A[M,K](bf16) * W[N,K]^T + biases ----------------
// 8 waves per block; wave tile 32x32; block tile 128x64.
__global__ void gemm_bf16_kernel(const __bf16* __restrict__ A, const __bf16* __restrict__ W,
                                 const float* __restrict__ bias1, const float* __restrict__ bias2,
                                 float* __restrict__ C, int M, int N, int K, int do_relu) {
  int lane = threadIdx.x & 31;
  int wid  = threadIdx.x >> 5;
  int m0 = blockIdx.x * 128 + (wid & 3) * 32;
  int n0 = blockIdx.y * 64  + (wid >> 2) * 32;
  f32x8 acc00 = {}, acc01 = {}, acc10 = {}, acc11 = {};
  int mr = lane & 15;
  for (int k0 = 0; k0 < K; k0 += 32) {
    bf16x16 a0 = fragA(A, m0 + mr,      K, k0, lane);
    bf16x16 a1 = fragA(A, m0 + 16 + mr, K, k0, lane);
    bf16x16 b0 = fragB(W, n0 + mr,      K, k0, lane);
    bf16x16 b1 = fragB(W, n0 + 16 + mr, K, k0, lane);
    acc00 = WMMA(a0, b0, acc00);
    acc01 = WMMA(a0, b1, acc01);
    acc10 = WMMA(a1, b0, acc10);
    acc11 = WMMA(a1, b1, acc11);
  }
  int hi = lane >> 4, nc = lane & 15;
#pragma unroll
  for (int mt = 0; mt < 2; ++mt) {
#pragma unroll
    for (int nt = 0; nt < 2; ++nt) {
      f32x8 acc = (mt == 0) ? (nt == 0 ? acc00 : acc01) : (nt == 0 ? acc10 : acc11);
#pragma unroll
      for (int r = 0; r < 8; ++r) {
        int m = m0 + mt * 16 + r + 8 * hi;
        int n = n0 + nt * 16 + nc;
        float v = acc[r];
        if (bias1) v += bias1[n];
        if (bias2) v += bias2[n];
        if (do_relu) v = fmaxf(v, 0.0f);
        C[(size_t)m * N + n] = v;
      }
    }
  }
}

// ---------------- elementwise f32 -> bf16 ----------------
__global__ void f2bf_kernel(const float* __restrict__ in, __bf16* __restrict__ out, size_t n) {
  for (size_t i = (size_t)blockIdx.x * blockDim.x + threadIdx.x; i < n;
       i += (size_t)gridDim.x * blockDim.x)
    out[i] = (__bf16)in[i];
}

// vT[b][h][d][s] = qkv_bf[b][s][1024 + h*64 + d]
__global__ void build_vt_kernel(const __bf16* __restrict__ qkv, __bf16* __restrict__ vT) {
  size_t total = (size_t)B_ * H_ * DH_ * S_;
  for (size_t idx = (size_t)blockIdx.x * blockDim.x + threadIdx.x; idx < total;
       idx += (size_t)gridDim.x * blockDim.x) {
    int s  = idx & (S_ - 1);
    int d  = (idx >> 10) & (DH_ - 1);
    int hh = (idx >> 16) & (H_ - 1);
    int b  = (int)(idx >> 19);
    vT[idx] = qkv[((size_t)b * S_ + s) * (3 * D_) + 2 * D_ + hh * DH_ + d];
  }
}

// hT[b][c][s] = (bf16) h[b][s][c]
__global__ void build_ht_kernel(const float* __restrict__ h, __bf16* __restrict__ hT) {
  size_t total = (size_t)B_ * D_ * S_;
  for (size_t idx = (size_t)blockIdx.x * blockDim.x + threadIdx.x; idx < total;
       idx += (size_t)gridDim.x * blockDim.x) {
    int s = idx & (S_ - 1);
    int c = (idx >> 10) & (D_ - 1);
    int b = (int)(idx >> 19);
    hT[idx] = (__bf16)h[((size_t)b * S_ + s) * D_ + c];
  }
}

// ---------------- BiLSTM recurrence (one 1024-thread WG per direction) ----------------
// per step: G[4,1024] = xp[:,t,:] + H[4,256] @ Whh^T (WMMA, M padded 4->16)
__global__ void lstm_kernel(const float* __restrict__ xp_f, const float* __restrict__ xp_b,
                            const __bf16* __restrict__ whh_f, const __bf16* __restrict__ whh_b,
                            float* __restrict__ hout) {
  int dir = blockIdx.x;
  const float*  xp  = dir ? xp_b  : xp_f;
  const __bf16* whh = dir ? whh_b : whh_f;
  int col_off = dir ? LH_ : 0;

  __shared__ __align__(16) __bf16 Hbf[16][LH_];   // A-matrix (rows 4..15 stay zero)
  __shared__ float cC[B_][LH_];
  __shared__ float G[B_][4 * LH_];

  for (int i = threadIdx.x; i < 16 * LH_; i += blockDim.x) ((__bf16*)Hbf)[i] = (__bf16)0.0f;
  for (int i = threadIdx.x; i < B_ * LH_; i += blockDim.x) ((float*)cC)[i] = 0.0f;
  __syncthreads();

  int lane = threadIdx.x & 31, wid = threadIdx.x >> 5;
  int n0 = wid * 32;
  int hi = lane >> 4, nc = lane & 15;

  for (int t = 0; t < S_; ++t) {
    int tt = dir ? (S_ - 1 - t) : t;
    f32x8 acc0 = {}, acc1 = {};
#pragma unroll
    for (int k0 = 0; k0 < LH_; k0 += 32) {
      bf16x16 a  = fragA((const __bf16*)Hbf, nc, LH_, k0, lane);
      bf16x16 b0 = fragB(whh, n0 + nc,      LH_, k0, lane);
      bf16x16 b1 = fragB(whh, n0 + 16 + nc, LH_, k0, lane);
      acc0 = WMMA(a, b0, acc0);
      acc1 = WMMA(a, b1, acc1);
    }
    if (hi == 0) {  // rows 0..3 are the real batches
#pragma unroll
      for (int r = 0; r < 4; ++r) {
        int n1 = n0 + nc, n2 = n0 + 16 + nc;
        const float* xr = xp + ((size_t)r * S_ + tt) * (4 * LH_);
        G[r][n1] = acc0[r] + xr[n1];
        G[r][n2] = acc1[r] + xr[n2];
      }
    }
    __syncthreads();
    {
      int b = threadIdx.x >> 8;      // 0..3
      int j = threadIdx.x & 255;     // 0..255
      float ig = G[b][j], fg = G[b][j + 256], gg = G[b][j + 512], og = G[b][j + 768];
      float c  = sigm(fg) * cC[b][j] + sigm(ig) * tanhf(gg);
      float hn = sigm(og) * tanhf(c);
      cC[b][j]  = c;
      Hbf[b][j] = (__bf16)hn;
      hout[((size_t)b * S_ + tt) * D_ + col_off + j] = hn;
    }
    __syncthreads();
  }
}

// ---------------- residual + LayerNorm (one 128-thread block per row) ----------------
__global__ void ln_kernel(float* __restrict__ x, const float* __restrict__ res,
                          const float* __restrict__ g, const float* __restrict__ bta,
                          __bf16* __restrict__ xbf) {
  int row = blockIdx.x, tid = threadIdx.x;
  size_t base = (size_t)row * D_ + tid * 4;
  f32x4 v = *(const f32x4*)(x + base);
  f32x4 rr = *(const f32x4*)(res + base);
  v.x += rr.x; v.y += rr.y; v.z += rr.z; v.w += rr.w;
  float s  = v.x + v.y + v.z + v.w;
  float ss = v.x * v.x + v.y * v.y + v.z * v.z + v.w * v.w;
  __shared__ float r1[128], r2[128];
  r1[tid] = s; r2[tid] = ss;
  __syncthreads();
  for (int o = 64; o > 0; o >>= 1) {
    if (tid < o) { r1[tid] += r1[tid + o]; r2[tid] += r2[tid + o]; }
    __syncthreads();
  }
  float mean = r1[0] * (1.0f / D_);
  float var  = r2[0] * (1.0f / D_) - mean * mean;
  float inv  = rsqrtf(var + 1e-5f);
#pragma unroll
  for (int k = 0; k < 4; ++k) {
    int c = tid * 4 + k;
    float vv = (k == 0 ? v.x : k == 1 ? v.y : k == 2 ? v.z : v.w);
    float o = (vv - mean) * inv * g[c] + bta[c];
    x[base + k]   = o;
    xbf[base + k] = (__bf16)o;
  }
}

// ---------------- flash-style attention (2-pass, WMMA everywhere) ----------------
// MODE 0: encoder MHA. qk = qkv_bf [B,S,1536]; bmat = vT [B*H][64][S];
//         out[b,i, h*64+d], NCOLS=64, scale=1/8, no rel bias.
// MODE 1: final rel-pos attention. qk = h_bf [B,S,512]; bmat = hT [B][512][S];
//         out[b,h,i,d] (d<512), NCOLS=512, scale=1, + rel_bias.
template <int MODE, int NCOLS>
__global__ void attn_kernel(const __bf16* __restrict__ qk, const __bf16* __restrict__ bmat,
                            const unsigned char* __restrict__ mask,
                            const float* __restrict__ rel_bias,
                            float* __restrict__ out, float scale) {
  __shared__ __align__(16) __bf16 Pw[8][16][32];  // per-wave P tile scratch
  int lane = threadIdx.x & 31, wid = threadIdx.x >> 5;
  int hi = lane >> 4, nc = lane & 15;
  int i0 = blockIdx.x * 128 + wid * 16;
  int bh = blockIdx.y;
  int b = bh / H_, h = bh % H_;

  const __bf16 *qbase, *kbase, *vbase;
  int qstride;
  if (MODE == 0) {
    qbase = qk + (size_t)b * S_ * (3 * D_) + h * DH_;
    kbase = qbase + D_;
    vbase = bmat + (size_t)bh * DH_ * S_;
    qstride = 3 * D_;
  } else {
    qbase = qk + (size_t)b * S_ * D_ + h * DH_;
    kbase = qbase;
    vbase = bmat + (size_t)b * D_ * S_;
    qstride = D_;
  }

  // Q fragments are reused for every key tile.
  bf16x16 aq0 = fragA(qbase, i0 + nc, qstride, 0, lane);
  bf16x16 aq1 = fragA(qbase, i0 + nc, qstride, 32, lane);

  // scores for a 16x16 tile at key offset j0 (C-layout: row = r + 8*hi, col = nc)
  auto score8 = [&](int j0, float sv[8]) {
    f32x8 s = {};
    s = WMMA(aq0, fragB(kbase, j0 + nc, qstride, 0, lane), s);
    s = WMMA(aq1, fragB(kbase, j0 + nc, qstride, 32, lane), s);
    int j = j0 + nc;
    float madd = mask[b * S_ + j] ? -1e9f : 0.0f;
#pragma unroll
    for (int r = 0; r < 8; ++r) {
      float v = s[r] * scale + madd;
      if (MODE == 1) {
        int i = i0 + r + 8 * hi;
        v += rel_bias[(size_t)(i - j + S_ - 1) * H_ + h];
      }
      sv[r] = v;
    }
  };

  // pass A: per-row running max / sum (rows live replicated across 16-lane halves)
  float mrow[8], lrow[8];
#pragma unroll
  for (int r = 0; r < 8; ++r) { mrow[r] = -1e30f; lrow[r] = 0.0f; }
  for (int j0 = 0; j0 < S_; j0 += 16) {
    float sv[8];
    score8(j0, sv);
#pragma unroll
    for (int r = 0; r < 8; ++r) {
      float x = sv[r];
      x = fmaxf(x, __shfl_xor(x, 1, 32));
      x = fmaxf(x, __shfl_xor(x, 2, 32));
      x = fmaxf(x, __shfl_xor(x, 4, 32));
      x = fmaxf(x, __shfl_xor(x, 8, 32));
      float mn = fmaxf(mrow[r], x);
      float e = expf(sv[r] - mn);
      e += __shfl_xor(e, 1, 32);
      e += __shfl_xor(e, 2, 32);
      e += __shfl_xor(e, 4, 32);
      e += __shfl_xor(e, 8, 32);
      lrow[r] = lrow[r] * expf(mrow[r] - mn) + e;
      mrow[r] = mn;
    }
  }
  float invl[8];
#pragma unroll
  for (int r = 0; r < 8; ++r) invl[r] = (lrow[r] > 0.0f) ? 1.0f / lrow[r] : 0.0f;

  // pass B: O[16, NCOLS] accumulated in CH-column chunks of registers
  constexpr int CH  = (NCOLS >= 128) ? 128 : NCOLS;
  constexpr int NTS = CH / 16;
  for (int c0 = 0; c0 < NCOLS; c0 += CH) {
    f32x8 acc[NTS] = {};
    for (int j0 = 0; j0 < S_; j0 += 32) {
#pragma unroll
      for (int half = 0; half < 2; ++half) {
        float sv[8];
        score8(j0 + half * 16, sv);
#pragma unroll
        for (int r = 0; r < 8; ++r) {
          float p = expf(sv[r] - mrow[r]) * invl[r];
          Pw[wid][r + 8 * hi][half * 16 + nc] = (__bf16)p;  // C-layout -> row-major
        }
      }
      asm volatile("s_wait_dscnt 0" ::: "memory");  // LDS stores -> loads (wave-local)
      bf16x16 ap = fragA(&Pw[wid][0][0], nc, 32, 0, lane);
#pragma unroll
      for (int dt = 0; dt < NTS; ++dt) {
        bf16x16 bv = fragB(vbase, c0 + dt * 16 + nc, S_, j0, lane);
        acc[dt] = WMMA(ap, bv, acc[dt]);
      }
    }
#pragma unroll
    for (int dt = 0; dt < NTS; ++dt) {
#pragma unroll
      for (int r = 0; r < 8; ++r) {
        int i = i0 + r + 8 * hi;
        int d = c0 + dt * 16 + nc;
        if (MODE == 0)
          out[((size_t)b * S_ + i) * D_ + h * DH_ + d] = acc[dt][r];
        else
          out[(((size_t)b * H_ + h) * S_ + i) * D_ + d] = acc[dt][r];
      }
    }
  }
}

// ---------------- workspace layout (bytes) ----------------
static constexpr size_t OFF_XBF   = 0;                                       // x bf16
static constexpr size_t OFF_WEMB  = OFF_XBF   + (size_t)B_ * S_ * IN_ * 2;
static constexpr size_t OFF_WIHF  = OFF_WEMB  + (size_t)D_ * IN_ * 2;
static constexpr size_t OFF_WIHB  = OFF_WIHF  + (size_t)4 * LH_ * D_ * 2;
static constexpr size_t OFF_WHHF  = OFF_WIHB  + (size_t)4 * LH_ * D_ * 2;
static constexpr size_t OFF_WHHB  = OFF_WHHF  + (size_t)4 * LH_ * LH_ * 2;
static constexpr size_t OFF_WQKV  = OFF_WHHB  + (size_t)4 * LH_ * LH_ * 2;
static constexpr size_t OFF_WO    = OFF_WQKV  + (size_t)L_ * 3 * D_ * D_ * 2;
static constexpr size_t OFF_W1    = OFF_WO    + (size_t)L_ * D_ * D_ * 2;
static constexpr size_t OFF_W2    = OFF_W1    + (size_t)L_ * FF_ * D_ * 2;
static constexpr size_t OFF_H     = OFF_W2    + (size_t)L_ * D_ * FF_ * 2;   // h f32
static constexpr size_t OFF_HBF   = OFF_H     + (size_t)B_ * S_ * D_ * 4;
static constexpr size_t OFF_BIG   = OFF_HBF   + (size_t)B_ * S_ * D_ * 2;    // 32MB: xp_f+xp_b | qkv f32 | ff1 f32
static constexpr size_t OFF_QKVBF = OFF_BIG   + (size_t)B_ * S_ * FF_ * 4;
static constexpr size_t OFF_VT    = OFF_QKVBF + (size_t)B_ * S_ * 3 * D_ * 2;
static constexpr size_t OFF_ATTN  = OFF_VT    + (size_t)B_ * H_ * DH_ * S_ * 2;  // attn/ff2 f32
static constexpr size_t OFF_ABF   = OFF_ATTN  + (size_t)B_ * S_ * D_ * 4;
static constexpr size_t OFF_FF1BF = OFF_ABF   + (size_t)B_ * S_ * D_ * 2;
static constexpr size_t OFF_HT    = OFF_FF1BF + (size_t)B_ * S_ * FF_ * 2;

extern "C" void kernel_launch(void* const* d_in, const int* in_sizes, int n_in,
                              void* d_out, int out_size, void* d_ws, size_t ws_size,
                              hipStream_t stream) {
  (void)in_sizes; (void)n_in; (void)out_size; (void)ws_size;
  char* ws = (char*)d_ws;
  auto bfp = [&](size_t off) { return (__bf16*)(ws + off); };
  auto fp  = [&](size_t off) { return (float*)(ws + off); };

  const float* x        = (const float*)d_in[0];
  const unsigned char* mask = (const unsigned char*)d_in[1];
  const float* emb_W    = (const float*)d_in[2];
  const float* emb_b    = (const float*)d_in[3];
  const float* Wih_f    = (const float*)d_in[4];
  const float* Whh_f    = (const float*)d_in[5];
  const float* bih_f    = (const float*)d_in[6];
  const float* bhh_f    = (const float*)d_in[7];
  const float* Wih_b    = (const float*)d_in[8];
  const float* Whh_b    = (const float*)d_in[9];
  const float* bih_b    = (const float*)d_in[10];
  const float* bhh_b    = (const float*)d_in[11];
  const float* Wqkv     = (const float*)d_in[12];
  const float* bqkv     = (const float*)d_in[13];
  const float* Wo       = (const float*)d_in[14];
  const float* bo       = (const float*)d_in[15];
  const float* ln1_g    = (const float*)d_in[16];
  const float* ln1_b    = (const float*)d_in[17];
  const float* ln2_g    = (const float*)d_in[18];
  const float* ln2_b    = (const float*)d_in[19];
  const float* W1       = (const float*)d_in[20];
  const float* b1       = (const float*)d_in[21];
  const float* W2       = (const float*)d_in[22];
  const float* b2       = (const float*)d_in[23];
  const float* rel_bias = (const float*)d_in[24];

  auto conv = [&](const float* src, __bf16* dst, size_t n) {
    f2bf_kernel<<<4096, 256, 0, stream>>>(src, dst, n);
  };
  auto gemm = [&](const __bf16* A, const __bf16* W, const float* bb1, const float* bb2,
                  float* C, int M, int N, int K, int relu) {
    dim3 g(M / 128, N / 64);
    gemm_bf16_kernel<<<g, 256, 0, stream>>>(A, W, bb1, bb2, C, M, N, K, relu);
  };

  const int MROWS = B_ * S_;  // 4096

  // ---- convert inputs/weights to bf16 (deterministic, every call) ----
  conv(x,     bfp(OFF_XBF),  (size_t)B_ * S_ * IN_);
  conv(emb_W, bfp(OFF_WEMB), (size_t)D_ * IN_);
  conv(Wih_f, bfp(OFF_WIHF), (size_t)4 * LH_ * D_);
  conv(Wih_b, bfp(OFF_WIHB), (size_t)4 * LH_ * D_);
  conv(Whh_f, bfp(OFF_WHHF), (size_t)4 * LH_ * LH_);
  conv(Whh_b, bfp(OFF_WHHB), (size_t)4 * LH_ * LH_);
  conv(Wqkv,  bfp(OFF_WQKV), (size_t)L_ * 3 * D_ * D_);
  conv(Wo,    bfp(OFF_WO),   (size_t)L_ * D_ * D_);
  conv(W1,    bfp(OFF_W1),   (size_t)L_ * FF_ * D_);
  conv(W2,    bfp(OFF_W2),   (size_t)L_ * D_ * FF_);

  // ---- embedding: h = x @ embW^T + emb_b ----
  gemm(bfp(OFF_XBF), bfp(OFF_WEMB), emb_b, nullptr, fp(OFF_H), MROWS, D_, IN_, 0);
  conv(fp(OFF_H), bfp(OFF_HBF), (size_t)MROWS * D_);

  // ---- BiLSTM ----
  float* xp_f = fp(OFF_BIG);
  float* xp_b = fp(OFF_BIG + (size_t)MROWS * 4 * LH_ * 4);
  gemm(bfp(OFF_HBF), bfp(OFF_WIHF), bih_f, bhh_f, xp_f, MROWS, 4 * LH_, D_, 0);
  gemm(bfp(OFF_HBF), bfp(OFF_WIHB), bih_b, bhh_b, xp_b, MROWS, 4 * LH_, D_, 0);
  lstm_kernel<<<2, 1024, 0, stream>>>(xp_f, xp_b, bfp(OFF_WHHF), bfp(OFF_WHHB), fp(OFF_H));
  conv(fp(OFF_H), bfp(OFF_HBF), (size_t)MROWS * D_);

  // ---- transformer encoder layers ----
  for (int l = 0; l < L_; ++l) {
    const __bf16* wq = bfp(OFF_WQKV) + (size_t)l * 3 * D_ * D_;
    const __bf16* wo = bfp(OFF_WO)   + (size_t)l * D_ * D_;
    const __bf16* w1 = bfp(OFF_W1)   + (size_t)l * FF_ * D_;
    const __bf16* w2 = bfp(OFF_W2)   + (size_t)l * D_ * FF_;

    // qkv projection
    gemm(bfp(OFF_HBF), wq, bqkv + (size_t)l * 3 * D_, nullptr, fp(OFF_BIG),
         MROWS, 3 * D_, D_, 0);
    conv(fp(OFF_BIG), bfp(OFF_QKVBF), (size_t)MROWS * 3 * D_);
    build_vt_kernel<<<4096, 256, 0, stream>>>(bfp(OFF_QKVBF), bfp(OFF_VT));

    // attention -> attn [B,S,D]
    attn_kernel<0, DH_><<<dim3(S_ / 128, B_ * H_), 256, 0, stream>>>(
        bfp(OFF_QKVBF), bfp(OFF_VT), mask, nullptr, fp(OFF_ATTN), 0.125f);

    // output projection (attn -> bf16 -> proj, proj reuses the attn f32 buffer)
    conv(fp(OFF_ATTN), bfp(OFF_ABF), (size_t)MROWS * D_);
    gemm(bfp(OFF_ABF), wo, bo + (size_t)l * D_, nullptr, fp(OFF_ATTN),
         MROWS, D_, D_, 0);

    // h = LN(h + proj)
    ln_kernel<<<MROWS, 128, 0, stream>>>(fp(OFF_H), fp(OFF_ATTN),
                                         ln1_g + (size_t)l * D_, ln1_b + (size_t)l * D_,
                                         bfp(OFF_HBF));

    // FFN: ff1 = relu(h W1^T + b1); ff2 = ff1 W2^T + b2
    gemm(bfp(OFF_HBF), w1, b1 + (size_t)l * FF_, nullptr, fp(OFF_BIG),
         MROWS, FF_, D_, 1);
    conv(fp(OFF_BIG), bfp(OFF_FF1BF), (size_t)MROWS * FF_);
    gemm(bfp(OFF_FF1BF), w2, b2 + (size_t)l * D_, nullptr, fp(OFF_ATTN),
         MROWS, D_, FF_, 0);

    // h = LN(h + ff2)
    ln_kernel<<<MROWS, 128, 0, stream>>>(fp(OFF_H), fp(OFF_ATTN),
                                         ln2_g + (size_t)l * D_, ln2_b + (size_t)l * D_,
                                         bfp(OFF_HBF));
  }

  // ---- final relative-position attention -> d_out [B,H,S,D] f32 ----
  build_ht_kernel<<<4096, 256, 0, stream>>>(fp(OFF_H), bfp(OFF_HT));
  attn_kernel<1, D_><<<dim3(S_ / 128, B_ * H_), 256, 0, stream>>>(
      bfp(OFF_HBF), bfp(OFF_HT), mask, rel_bias, (float*)d_out, 1.0f);
}
